// LovaszSoftmaxLoss_36051955483255
// MI455X (gfx1250) — matrix operand order repaired and checked by reading, not compiled, not run on hardware
//
#include <hip/hip_runtime.h>
#include <stdint.h>

// ---------------- problem constants (from reference) ----------------
#define C_CLS 19
#define B_    8
#define H_    512
#define W_    512
#define HW    (H_*W_)            // 262144
#define N_PIX (B_*HW)            // 2097152 pixels
#define TPB   256                // 8 wave32 per block
#define EPT   16                 // elements per thread
#define TILE  (TPB*EPT)          // 4096 keys per sort block
#define NBLK  (N_PIX/TILE)       // 512 sort blocks per class

typedef unsigned int u32;
typedef _Float16 h16 __attribute__((ext_vector_type(16)));
typedef float    f8  __attribute__((ext_vector_type(8)));

#define USE_ASYNC_LDS 1

__device__ __forceinline__ u32 ballot32(bool p) {
#if defined(__has_builtin)
#if __has_builtin(__builtin_amdgcn_ballot_w32)
  return __builtin_amdgcn_ballot_w32(p);
#else
  return (u32)__ballot(p);
#endif
#else
  return (u32)__ballot(p);
#endif
}

// ============================================================================
// K1: softmax over 19 classes per pixel, emit sortable keys per class.
// key = ~((bits(e)<<1) | is_target); e in [0,1] so bits(e) < 2^30 -> fits.
// Ascending radix sort of key == descending e (payload bit rides along).
// ============================================================================
__global__ void k1_errors(const float* __restrict__ pred,
                          const int* __restrict__ target,
                          u32* __restrict__ keysA, size_t clsStride,
                          int g0, int g) {
  int n = blockIdx.x * TPB + threadIdx.x;          // grid.x = N_PIX/TPB exactly
  int b = n / HW, p = n % HW;
  const float* base = pred + (size_t)b * C_CLS * HW + p;
  float x[C_CLS];
#pragma unroll
  for (int c = 0; c < C_CLS; ++c) x[c] = base[(size_t)c * HW];
  float m = x[0];
#pragma unroll
  for (int c = 1; c < C_CLS; ++c) m = fmaxf(m, x[c]);
  float s = 0.f;
#pragma unroll
  for (int c = 0; c < C_CLS; ++c) { x[c] = expf(x[c] - m); s += x[c]; }
  float inv = 1.0f / s;
  // Harness doc: integer inputs are passed as const int* (labels 0..18).
  int t = target[n];
#pragma unroll
  for (int c = 0; c < C_CLS; ++c) {
    int cg = c - g0;
    if (cg >= 0 && cg < g) {
      float pc = x[c] * inv;
      u32 isT = (t == c) ? 1u : 0u;
      float e = isT ? (1.0f - pc) : pc;
      u32 key = ~((__float_as_uint(e) << 1) | isT);
      keysA[(size_t)cg * clsStride + n] = key;
    }
  }
}

// ============================================================================
// K2: radix histogram. Tile staged into LDS with CDNA5 async global->LDS DMA,
// 16 bytes per lane per issue (B128), 512 contiguous bytes per wave issue.
// ============================================================================
__global__ void k2_hist(const u32* __restrict__ keysIn, u32* __restrict__ cnts,
                        size_t clsStride, int shift) {
  int cg = blockIdx.y, blk = blockIdx.x, t = threadIdx.x;
  const u32* in = keysIn + (size_t)cg * clsStride + (size_t)blk * TILE;
  u32* cnt = cnts + (size_t)cg * clsStride + (size_t)blk * 256;
  __shared__ uint4 stage4[TILE / 4];     // uint4 -> 16B alignment for B128
  __shared__ u32 hist[256];
  u32* stage = (u32*)stage4;
  hist[t] = 0;
#if USE_ASYNC_LDS
#pragma unroll
  for (int q = 0; q < 4; ++q) {
    // LDS byte address = low 32 bits of the generic shared address (ISA 10.2).
    unsigned lds = (unsigned)(uintptr_t)&stage4[q * TPB + t];
    unsigned long long ga =
        (unsigned long long)(uintptr_t)(in + (size_t)(q * TPB + t) * 4);
    asm volatile("global_load_async_to_lds_b128 %0, %1, off"
                 :: "v"(lds), "v"(ga) : "memory");
  }
  asm volatile("s_wait_asynccnt 0" ::: "memory");
#else
#pragma unroll
  for (int i = 0; i < EPT; ++i) stage[i * TPB + t] = in[i * TPB + t];
#endif
  __syncthreads();
#pragma unroll
  for (int i = 0; i < EPT; ++i)
    atomicAdd(&hist[(stage[i * TPB + t] >> shift) & 255u], 1u);
  __syncthreads();
  cnt[t] = hist[t];
}

// ============================================================================
// K3: per class, turn per-block digit counts into global scatter bases
// (digit-major exclusive prefix). One block (256 threads) per class.
// ============================================================================
__global__ void k3_scan(u32* __restrict__ cnts, size_t clsStride) {
  int cg = blockIdx.x, t = threadIdx.x;
  u32* cnt = cnts + (size_t)cg * clsStride;
  u32 total = 0;
  for (int blk = 0; blk < NBLK; ++blk) total += cnt[blk * 256 + t];
  __shared__ u32 sc[256];
  sc[t] = total; __syncthreads();
  for (int off = 1; off < 256; off <<= 1) {
    u32 v = (t >= off) ? sc[t - off] : 0u; __syncthreads();
    sc[t] += v; __syncthreads();
  }
  u32 run = sc[t] - total;                 // exclusive digit start
  for (int blk = 0; blk < NBLK; ++blk) {
    u32 v = cnt[blk * 256 + t];
    cnt[blk * 256 + t] = run;              // becomes this block's base for digit t
    run += v;
  }
}

// ============================================================================
// K4: stable scatter. Ranks are built with wave32 ballots (deterministic,
// no LDS-atomic order races), staged through LDS so global writes are
// coalesced runs per digit.
// ============================================================================
__global__ void k4_scatter(const u32* __restrict__ keysIn, u32* __restrict__ keysOut,
                           const u32* __restrict__ cnts, size_t clsStride, int shift) {
  int cg = blockIdx.y, blk = blockIdx.x, t = threadIdx.x;
  int lane = t & 31, wave = t >> 5;
  const u32* in = keysIn + (size_t)cg * clsStride + (size_t)blk * TILE;
  u32* out = keysOut + (size_t)cg * clsStride;
  const u32* cnt = cnts + (size_t)cg * clsStride + (size_t)blk * 256;
  __shared__ u32 stage[TILE];
  __shared__ u32 waveHist[8 * 256];
  __shared__ u32 hist[256], localStart[256], running[256], sBase[256], sc[256];
  u32 k[EPT];
#pragma unroll
  for (int i = 0; i < EPT; ++i) k[i] = in[i * TPB + t];   // element order = i*256+t
  hist[t] = 0; running[t] = 0; sBase[t] = cnt[t];
  __syncthreads();
#pragma unroll
  for (int i = 0; i < EPT; ++i) atomicAdd(&hist[(k[i] >> shift) & 255u], 1u);
  __syncthreads();
  u32 hv = hist[t];
  sc[t] = hv; __syncthreads();
  for (int off = 1; off < 256; off <<= 1) {
    u32 v = (t >= off) ? sc[t - off] : 0u; __syncthreads();
    sc[t] += v; __syncthreads();
  }
  localStart[t] = sc[t] - hv;
  __syncthreads();
  for (int i = 0; i < EPT; ++i) {
    u32 d = (k[i] >> shift) & 255u;
#pragma unroll
    for (int w = 0; w < 8; ++w) waveHist[w * 256 + t] = 0;
    __syncthreads();
    // wave32 multi-split: mask of lanes sharing my digit
    u32 mask = 0xFFFFFFFFu;
#pragma unroll
    for (int bit = 0; bit < 8; ++bit) {
      u32 bal = ballot32(((d >> bit) & 1u) != 0u);
      mask &= ((d >> bit) & 1u) ? bal : ~bal;
    }
    u32 peerRank = __popc(mask & ((1u << lane) - 1u));
    if (peerRank == 0) waveHist[wave * 256 + d] = __popc(mask);
    __syncthreads();
    u32 cross = 0;
    for (int w = 0; w < wave; ++w) cross += waveHist[w * 256 + d];
    u32 pos = localStart[d] + running[d] + cross + peerRank;   // stable rank
    stage[pos] = k[i];
    __syncthreads();
    u32 tot = 0;
#pragma unroll
    for (int w = 0; w < 8; ++w) tot += waveHist[w * 256 + t];
    running[t] += tot;
    __syncthreads();
  }
#pragma unroll
  for (int i = 0; i < EPT; ++i) {
    int j = i * TPB + t;
    u32 key = stage[j];
    u32 d = (key >> shift) & 255u;
    out[sBase[d] + (u32)j - localStart[d]] = key;   // coalesced per digit run
  }
}

// ============================================================================
// K5: per-tile count of target bits (b = LSB of ~key) in sorted order.
// ============================================================================
__global__ void k5_chunksum(const u32* __restrict__ keys, size_t clsStride,
                            u32* __restrict__ chunkBuf, int g0) {
  int cg = blockIdx.y, blk = blockIdx.x, t = threadIdx.x;
  const u32* in = keys + (size_t)cg * clsStride + (size_t)blk * TILE;
  u32 s = 0;
#pragma unroll
  for (int i = 0; i < EPT; ++i) s += (~in[i * TPB + t]) & 1u;
  __shared__ u32 red[256];
  red[t] = s; __syncthreads();
  for (int off = 128; off > 0; off >>= 1) {
    if (t < off) red[t] += red[t + off];
    __syncthreads();
  }
  if (t == 0) chunkBuf[(size_t)(g0 + cg) * NBLK + blk] = red[0];
}

// ============================================================================
// K6: exclusive scan of the 512 chunk sums per class; total = gts[c].
// ============================================================================
__global__ void k6_scanchunks(u32* __restrict__ chunkBuf, u32* __restrict__ gts, int g0) {
  int cg = blockIdx.x, t = threadIdx.x;                 // blockDim.x == NBLK
  u32* cb = chunkBuf + (size_t)(g0 + cg) * NBLK;
  u32 v = cb[t];
  __shared__ u32 sc[NBLK];
  sc[t] = v; __syncthreads();
  for (int off = 1; off < NBLK; off <<= 1) {
    u32 x = (t >= off) ? sc[t - off] : 0u; __syncthreads();
    sc[t] += x; __syncthreads();
  }
  cb[t] = sc[t] - v;
  if (t == NBLK - 1) gts[g0 + cg] = sc[t];
}

// ============================================================================
// K7: Lovasz gradient + dot product over the sorted order.
// Block-wide prefix of 256 thread bit-sums is computed on the MATRIX pipe:
// inclusive row prefix = M(16x16) x U(upper-triangular) via one
// v_wmma_f32_16x16x32_f16 (thread sums <= 16 exact in f16, f32 accumulate).
// ============================================================================
__global__ void k7_reduce(const u32* __restrict__ keys, size_t clsStride,
                          const u32* __restrict__ chunkBuf, const u32* __restrict__ gts,
                          float* __restrict__ partials, int g0) {
  int cg = blockIdx.y, blk = blockIdx.x, t = threadIdx.x;
  int c = g0 + cg;
  const u32* base = keys + (size_t)cg * clsStride + (size_t)blk * TILE;
  const uint4* v4 = (const uint4*)base;
  u32 kk[EPT];
#pragma unroll
  for (int q = 0; q < 4; ++q) {
    uint4 v = v4[t * 4 + q];
    kk[q * 4 + 0] = v.x; kk[q * 4 + 1] = v.y; kk[q * 4 + 2] = v.z; kk[q * 4 + 3] = v.w;
  }
  u32 bsum = 0;
#pragma unroll
  for (int j = 0; j < EPT; ++j) bsum += (~kk[j]) & 1u;
  __shared__ float sSum[256];
  __shared__ float incl[256];
  sSum[t] = (float)bsum;
  __syncthreads();
  if (t < 32) {                               // wave 0, EXEC all ones
    h16 a, bm; f8 c0 = {};
    int mrow = (t < 16) ? t : (t - 16);
    int kbase = (t < 16) ? 0 : 8;             // A 16x32 f16 layout (K>=16 zero-padded)
#pragma unroll
    for (int h = 0; h < 8; ++h) a[h] = (_Float16)sSum[mrow * 16 + kbase + h];
#pragma unroll
    for (int h = 8; h < 16; ++h) a[h] = (_Float16)0.f;
#pragma unroll
    for (int h = 0; h < 16; ++h) {            // B 32x16: rows K=kbase+h (h<8), col=lane%16
      float uv = (h < 8 && kbase + h <= mrow) ? 1.f : 0.f;   // upper-tri incl diagonal
      bm[h] = (_Float16)uv;
    }
    f8 d = __builtin_amdgcn_wmma_f32_16x16x32_f16(false, a, false, bm,
                                                  (short)0, c0, false, false);
    int hi = (t < 16) ? 0 : 8, col = t & 15;
#pragma unroll
    for (int v = 0; v < 8; ++v) incl[(v + hi) * 16 + col] = d[v];
  }
  __syncthreads();
  int r = t >> 4;
  float rowOff = 0.f;
  for (int rr = 0; rr < r; ++rr) rowOff += incl[rr * 16 + 15];
  u32 excl = (u32)(rowOff + incl[t] - sSum[t] + 0.5f);   // exact small ints
  float acc = 0.f;
  u32 gtsv = gts[c];
  if (gtsv > 0) {
    float gtsF = (float)gtsv;
    u32 cumB = chunkBuf[(size_t)c * NBLK + blk] + excl;
    int idx0 = blk * TILE + t * EPT;
#pragma unroll
    for (int j = 0; j < EPT; ++j) {
      u32 v = ~kk[j];
      u32 b = v & 1u;
      float e = __uint_as_float(v >> 1);
      cumB += b;
      int idx = idx0 + j;
      float fi = (float)idx;                  // < 2^24, exact
      float cb = (float)cumB;
      float J = 1.f - (gtsF - cb) / (gtsF + (fi + 1.f) - cb);
      float Jp = 0.f;
      if (idx > 0) {
        float cbp = cb - (float)b;
        Jp = 1.f - (gtsF - cbp) / (gtsF + fi - cbp);
      }
      acc += (J - Jp) * e;
    }
  }
  __shared__ float red[256];
  red[t] = acc; __syncthreads();
  for (int off = 128; off > 0; off >>= 1) {
    if (t < off) red[t] += red[t + off];
    __syncthreads();
  }
  if (t == 0) partials[(size_t)c * NBLK + blk] = red[0];  // fixed-order: deterministic
}

// ============================================================================
// K8: final deterministic reduce across blocks and classes.
// ============================================================================
__global__ void k8_final(const float* __restrict__ partials, const u32* __restrict__ gts,
                         float* __restrict__ out) {
  int t = threadIdx.x;
  __shared__ float red[256];
  float lossSum = 0.f; int np = 0;
  for (int c = 0; c < C_CLS; ++c) {
    float v = partials[(size_t)c * NBLK + t] + partials[(size_t)c * NBLK + 256 + t];
    red[t] = v; __syncthreads();
    for (int off = 128; off > 0; off >>= 1) {
      if (t < off) red[t] += red[t + off];
      __syncthreads();
    }
    if (t == 0 && gts[c] > 0) { lossSum += red[0]; np++; }
    __syncthreads();
  }
  if (t == 0) out[0] = lossSum / (float)(np > 0 ? np : 1);
}

// ============================================================================
// Host: class-grouped pipeline sized to ws_size (~17.3 MB per class).
// ============================================================================
extern "C" void kernel_launch(void* const* d_in, const int* in_sizes, int n_in,
                              void* d_out, int out_size, void* d_ws, size_t ws_size,
                              hipStream_t stream) {
  (void)in_sizes; (void)n_in; (void)out_size;
  const float* pred = (const float*)d_in[0];
  const int* target = (const int*)d_in[1];
  float* out = (float*)d_out;
  unsigned char* ws = (unsigned char*)d_ws;

  size_t chunkOff = 0;
  size_t partOff  = chunkOff + (size_t)C_CLS * NBLK * 4;
  size_t gtsOff   = partOff  + (size_t)C_CLS * NBLK * 4;
  size_t fixedEnd = (gtsOff + (size_t)C_CLS * 4 + 255) & ~(size_t)255;
  size_t perClass = (size_t)N_PIX * 8 + (size_t)NBLK * 256 * 4;  // keysA+keysB+counts
  size_t clsStride = perClass / 4;                                // in u32 words

  size_t avail = ws_size > fixedEnd ? ws_size - fixedEnd : 0;
  int G = (int)(avail / perClass);
  if (G < 1) G = 1;
  if (G > C_CLS) G = C_CLS;

  u32* chunkBuf  = (u32*)(ws + chunkOff);
  float* partials = (float*)(ws + partOff);
  u32* gts = (u32*)(ws + gtsOff);
  u32* cls0 = (u32*)(ws + fixedEnd);

  for (int g0 = 0; g0 < C_CLS; g0 += G) {
    int g = (C_CLS - g0 < G) ? (C_CLS - g0) : G;
    u32* keysA = cls0;                         // per-class offset = cg*clsStride
    u32* keysB = cls0 + (size_t)N_PIX;
    u32* cnts  = cls0 + 2 * (size_t)N_PIX;

    k1_errors<<<N_PIX / TPB, TPB, 0, stream>>>(pred, target, keysA, clsStride, g0, g);

    u32 *src = keysA, *dst = keysB;
    for (int p = 0; p < 4; ++p) {              // LSD radix, ends back in keysA
      k2_hist   <<<dim3(NBLK, g), TPB, 0, stream>>>(src, cnts, clsStride, 8 * p);
      k3_scan   <<<g, 256, 0, stream>>>(cnts, clsStride);
      k4_scatter<<<dim3(NBLK, g), TPB, 0, stream>>>(src, dst, cnts, clsStride, 8 * p);
      u32* tmp = src; src = dst; dst = tmp;
    }
    k5_chunksum <<<dim3(NBLK, g), TPB, 0, stream>>>(src, clsStride, chunkBuf, g0);
    k6_scanchunks<<<g, NBLK, 0, stream>>>(chunkBuf, gts, g0);
    k7_reduce   <<<dim3(NBLK, g), TPB, 0, stream>>>(src, clsStride, chunkBuf, gts,
                                                    partials, g0);
  }
  k8_final<<<1, 256, 0, stream>>>(partials, gts, out);
}
